// PGAHead_87127706566938
// MI455X (gfx1250) — matrix operand order repaired and checked by reading, not compiled, not run on hardware
//
#include <hip/hip_runtime.h>

// ---------------- constants ----------------
#define B_FULL 2048
#define DFEAT  512
#define NLAYER 8
#define KTOP   5

typedef _Float16 v16h __attribute__((ext_vector_type(16)));
typedef _Float16 v8h  __attribute__((ext_vector_type(8)));
typedef float    v8f  __attribute__((ext_vector_type(8)));

// ---------------- workspace layout (all offsets multiple of 256) ----------------
constexpr size_t OFF_META  = 0;                        // meta[0] = Bsub
constexpr size_t OFF_IDX   = 256;                      // 2048 int
constexpr size_t OFF_LSUB  = OFF_IDX  + 8192;          // 2048 int
constexpr size_t OFF_FLAGS = OFF_LSUB + 8192;          // 2048 int
constexpr size_t OFF_KEEP  = OFF_FLAGS+ 8192;          // 2048 u8
constexpr size_t OFF_TOPI  = OFF_KEEP + 2048;          // 2048*5 int
constexpr size_t OFF_EMA0  = OFF_TOPI + 40960;         // 2048 f
constexpr size_t OFF_EMA1  = OFF_EMA0 + 8192;
constexpr size_t OFF_PNUM  = OFF_EMA1 + 8192;
constexpr size_t OFF_PDEN  = OFF_PNUM + 8192;
constexpr size_t OFF_PCNT  = OFF_PDEN + 8192;
constexpr size_t OFF_TERM  = OFF_PCNT + 8192;          // 7 f
constexpr size_t OFF_XH    = OFF_TERM + 256;           // 2048*512 f16 = 2MB
constexpr size_t OFF_S0    = OFF_XH   + 2097152;       // 2048*2048 f32 = 16MB
constexpr size_t OFF_S1    = OFF_S0   + 16777216;
constexpr size_t OFF_M0    = OFF_S1   + 16777216;      // 2048*2048 u8 = 4MB
constexpr size_t OFF_M1    = OFF_M0   + 4194304;

// ---------------- helpers ----------------
__device__ __forceinline__ float warp_sum_f(float v) {
  for (int o = 16; o; o >>= 1) v += __shfl_xor(v, o, 32);
  return v;
}
__device__ __forceinline__ int warp_sum_i(int v) {
  for (int o = 16; o; o >>= 1) v += __shfl_xor(v, o, 32);
  return v;
}

// ---------------- 1) normalize (+optional gather) to f16 ----------------
// one wave32 per row; rows >= count are zero-filled
__global__ void normalize_gather(const float* __restrict__ feats,
                                 const int* __restrict__ idx_map,
                                 const int* __restrict__ countPtr,
                                 _Float16* __restrict__ Xh) {
  int wave = threadIdx.x >> 5, lane = threadIdx.x & 31;
  int r = blockIdx.x * 8 + wave;
  int count = countPtr ? *countPtr : B_FULL;
  const float* src = nullptr;
  if (r < count) {
    int s = idx_map ? idx_map[r] : r;
    src = feats + (size_t)s * DFEAT;
  }
  float ss = 0.0f;
  if (src) for (int j = lane; j < DFEAT; j += 32) { float v = src[j]; ss += v * v; }
  ss = warp_sum_f(ss);
  float inv = src ? (1.0f / fmaxf(sqrtf(ss), 1e-8f)) : 0.0f;
  for (int j = lane; j < DFEAT; j += 32) {
    float v = src ? src[j] * inv : 0.0f;
    Xh[(size_t)r * DFEAT + j] = (_Float16)v;
  }
}

// ---------------- 2) S = clip(Xn @ Xn^T) via v_wmma_f32_16x16x32_f16 ----------------
// block = 128 threads (4 waves), each wave computes one 16x16 tile
__global__ void gemm_xxt(const _Float16* __restrict__ X, float* __restrict__ S) {
  int wave = threadIdx.x >> 5, lane = threadIdx.x & 31;
  int tn = (blockIdx.x * 4 + wave) * 16;   // tile col base
  int tm = blockIdx.y * 16;                // tile row base
  int lo = lane & 15, hi = lane >> 4;
  const _Float16* arow = X + (size_t)(tm + lo) * DFEAT + hi * 16;  // A: 16x32 layout
  const _Float16* brow = X + (size_t)(tn + lo) * DFEAT + hi * 16;  // B: 32x16 layout (cols = rows of X)
  v8f c = {};
  for (int k = 0; k < DFEAT; k += 32) {
    v8h a0 = *(const v8h*)(arow + k);
    v8h a1 = *(const v8h*)(arow + k + 8);
    v16h a;
#pragma unroll
    for (int t = 0; t < 8; ++t) { a[t] = a0[t]; a[t + 8] = a1[t]; }
    v16h b = *(const v16h*)(brow + k);
    c = __builtin_amdgcn_wmma_f32_16x16x32_f16(false, a, false, b, (short)0, c,
                                               false, false);
  }
#pragma unroll
  for (int r = 0; r < 8; ++r) {
    float v = c[r];
    v = fminf(fmaxf(v, -1.0f + 1e-8f), 1.0f - 1e-8f);
    S[(size_t)(tm + r + 8 * hi) * B_FULL + tn + lo] = v;
  }
}

// ---------------- 3) per-row masked top-5 + candidate count ----------------
// one wave per row; merge via LDS, lane 0 does 5 selection passes (deterministic)
__global__ void topk_rows(const float* __restrict__ S, const int* __restrict__ lab,
                          const int* __restrict__ countPtr,
                          int* __restrict__ topidx, unsigned char* __restrict__ keep) {
  __shared__ float sv[8][160];
  __shared__ int   si[8][160];
  int wave = threadIdx.x >> 5, lane = threadIdx.x & 31;
  int row = blockIdx.x * 8 + wave;
  int count = countPtr ? *countPtr : B_FULL;
  float bv[KTOP]; int bi[KTOP];
#pragma unroll
  for (int t = 0; t < KTOP; ++t) { bv[t] = -3.0e38f; bi[t] = -1; }
  int cand = 0;
  if (row < count) {
    int myLab = lab[row];
    const float* srow = S + (size_t)row * B_FULL;
    for (int j = lane; j < count; j += 32) {
      float m;
      if (j == row || lab[j] != myLab) m = -1.0e9f;
      else { float s = srow[j]; m = (s < 0.0f) ? -1.0e9f : s; }
      if (m > 0.0f) cand++;
#pragma unroll
      for (int t = 0; t < KTOP; ++t) {
        if (m > bv[t] || (m == bv[t] && j < bi[t])) {
          for (int u = KTOP - 1; u > t; --u) { bv[u] = bv[u - 1]; bi[u] = bi[u - 1]; }
          bv[t] = m; bi[t] = j;
          break;
        }
      }
    }
  }
  cand = warp_sum_i(cand);
#pragma unroll
  for (int t = 0; t < KTOP; ++t) { sv[wave][lane * KTOP + t] = bv[t]; si[wave][lane * KTOP + t] = bi[t]; }
  __syncthreads();
  if (lane == 0) {
    if (row < count) {
      keep[row] = (cand >= KTOP) ? 1 : 0;
      for (int k = 0; k < KTOP; ++k) {
        float best = -3.2e38f; int besti = 0x7fffffff; int slot = -1;
        for (int m = 0; m < 160; ++m) {
          float v = sv[wave][m]; int idv = si[wave][m];
          if (v > best || (v == best && idv < besti)) { best = v; besti = idv; slot = m; }
        }
        topidx[row * KTOP + k] = (best > -2.9e38f) ? besti : -1;
        sv[wave][slot] = -3.4e38f; si[wave][slot] = 0x7fffffff;
      }
    } else if (row < B_FULL) {
      keep[row] = 0;
    }
  }
}

// ---------------- 4) build symmetric kNN mask M (full 2048x2048, zeros out of range) ----
__global__ void build_M(const int* __restrict__ lab, const int* __restrict__ countPtr,
                        const int* __restrict__ topidx, const unsigned char* __restrict__ keep,
                        unsigned char* __restrict__ M) {
  __shared__ int rT[80], cT[80];
  __shared__ int rL[16], cL[16];
  __shared__ unsigned char rK[16], cK[16];
  int count = countPtr ? *countPtr : B_FULL;
  int tx = threadIdx.x, ty = threadIdx.y;
  int t = ty * 16 + tx;
  int i0 = blockIdx.y * 16, j0 = blockIdx.x * 16;
  if (t < 16) {
    int ii = i0 + t;
    rL[t] = (ii < count) ? lab[ii] : -1;
    rK[t] = (ii < count) ? keep[ii] : (unsigned char)0;
    int jj = j0 + t;
    cL[t] = (jj < count) ? lab[jj] : -2;
    cK[t] = (jj < count) ? keep[jj] : (unsigned char)0;
  }
  if (t < 80) {
    int ii = i0 + t / KTOP;
    rT[t] = (ii < count) ? topidx[ii * KTOP + (t % KTOP)] : -1;
    int jj = j0 + t / KTOP;
    cT[t] = (jj < count) ? topidx[jj * KTOP + (t % KTOP)] : -1;
  }
  __syncthreads();
  int i = i0 + ty, j = j0 + tx;
  unsigned char m = 0;
  if (i < count && j < count && i != j && rL[ty] == cL[tx] && rK[ty] && cK[tx]) {
    bool in = false;
#pragma unroll
    for (int k = 0; k < KTOP; ++k) in = in || (rT[ty * KTOP + k] == j) || (cT[tx * KTOP + k] == i);
    m = in ? 1 : 0;
  }
  M[(size_t)i * B_FULL + j] = m;
}

// ---------------- 5) row-any over M (candidate rows) ----------------
__global__ void row_any(const unsigned char* __restrict__ M, int* __restrict__ flags) {
  int wave = threadIdx.x >> 5, lane = threadIdx.x & 31;
  int row = blockIdx.x * 8 + wave;
  const unsigned int* w = (const unsigned int*)(M + (size_t)row * B_FULL);
  int acc = 0;
  for (int j = lane; j < B_FULL / 4; j += 32) acc |= (int)(w[j] != 0u);
  acc = warp_sum_i(acc);
  if (lane == 0) flags[row] = (acc != 0) ? 1 : 0;
}

// ---------------- 6) compaction (single thread, deterministic) ----------------
__global__ void compact(const int* __restrict__ flags, const int* __restrict__ labels,
                        int* __restrict__ meta, int* __restrict__ idx_sub,
                        int* __restrict__ lsub) {
  int c = 0;
  for (int i = 0; i < B_FULL; ++i)
    if (flags[i]) { idx_sub[c] = i; lsub[c] = labels[i]; ++c; }
  meta[0] = c;
}

// ---------------- 7) degree/sigmoid EMA score per row ----------------
__global__ void score_rows(const float* __restrict__ S, const int* __restrict__ lab,
                           const int* __restrict__ countPtr, float* __restrict__ ema) {
  int wave = threadIdx.x >> 5, lane = threadIdx.x & 31;
  int row = blockIdx.x * 8 + wave;
  int count = *countPtr;
  float sum = 0.0f; int deg = 0;
  if (row < count) {
    int myLab = lab[row];
    const float* sr = S + (size_t)row * B_FULL;
    for (int j = lane; j < count; j += 32) {
      if (lab[j] == myLab) { float s = sr[j]; if (s > 0.0f) { sum += s; deg++; } }
    }
  }
  sum = warp_sum_f(sum);
  deg = warp_sum_i(deg);
  if (lane == 0 && row < B_FULL) {
    float d = fmaxf((float)deg, 1.0f);
    float sc = 1.0f / (1.0f + expf(-(sum / d)));
    ema[row] = (row < count) ? (0.45f + 0.1f * sc) : 0.0f;  // MOM*0.5 + (1-MOM)*score
  }
}

// ---------------- 8) per-pair row partials (deterministic LDS tree) ----------------
__global__ void pair_rows(const float* __restrict__ Sp, const float* __restrict__ Sc,
                          const unsigned char* __restrict__ Mp, const unsigned char* __restrict__ Mc,
                          const float* __restrict__ emap, const int* __restrict__ countPtr,
                          float* __restrict__ pnum, float* __restrict__ pden,
                          float* __restrict__ pcnt) {
  __shared__ float sn[256], sd[256], sc2[256];
  int i = blockIdx.x;
  int t = threadIdx.x;
  int count = *countPtr;
  float n = 0.0f, d = 0.0f, c = 0.0f;
  if (i < count) {
    float ei = emap[i];
    const float* sp = Sp + (size_t)i * B_FULL;
    const float* su = Sc + (size_t)i * B_FULL;
    const unsigned char* mp = Mp + (size_t)i * B_FULL;
    const unsigned char* mc = Mc + (size_t)i * B_FULL;
    for (int j = t; j < count; j += 256) {
      unsigned char mpj = mp[j], mcj = mc[j];
      if (mpj | mcj) {
        float w = ei * emap[j];
        float at = mpj ? fmaxf(sp[j], 0.0f) : 0.0f;  // A_tgt  = max(sim,0)*M
        float ad = mcj ? fmaxf(su[j], 0.0f) : 0.0f;  // A_deep
        float df = ad - at;
        n += df * df * w;
        d += w;
        c += 1.0f;
      }
    }
  }
  sn[t] = n; sd[t] = d; sc2[t] = c;
  __syncthreads();
  for (int s = 128; s; s >>= 1) {
    if (t < s) { sn[t] += sn[t + s]; sd[t] += sd[t + s]; sc2[t] += sc2[t + s]; }
    __syncthreads();
  }
  if (t == 0) { pnum[i] = sn[0]; pden[i] = sd[0]; pcnt[i] = sc2[0]; }
}

__global__ void pair_reduce(const float* __restrict__ pnum, const float* __restrict__ pden,
                            const float* __restrict__ pcnt, float* __restrict__ term, int p) {
  float n = 0.0f, d = 0.0f, c = 0.0f;
  for (int i = 0; i < B_FULL; ++i) { n += pnum[i]; d += pden[i]; c += pcnt[i]; }
  term[p] = (c > 0.0f) ? (n / fmaxf(d, 1e-8f)) : 0.0f;
}

__global__ void finalize(const float* __restrict__ term, float* __restrict__ out) {
  float s = 0.0f;
  for (int p = 0; p < NLAYER - 1; ++p) s += term[p];
  out[0] = 16.0f * (s / (float)(NLAYER - 1));  // LAMBDA_ALIGN_K * raw
}

// ---------------- launch ----------------
extern "C" void kernel_launch(void* const* d_in, const int* in_sizes, int n_in,
                              void* d_out, int out_size, void* d_ws, size_t ws_size,
                              hipStream_t stream) {
  const float* F      = (const float*)d_in[0];  // (8, 2048, 512)
  const int*   labels = (const int*)d_in[1];    // (2048,)
  (void)d_in; (void)in_sizes; (void)n_in; (void)out_size; (void)ws_size;

  char* ws = (char*)d_ws;
  int*            meta    = (int*)(ws + OFF_META);
  int*            idx_sub = (int*)(ws + OFF_IDX);
  int*            lsub    = (int*)(ws + OFF_LSUB);
  int*            flags   = (int*)(ws + OFF_FLAGS);
  unsigned char*  keep    = (unsigned char*)(ws + OFF_KEEP);
  int*            topi    = (int*)(ws + OFF_TOPI);
  float*          ema0    = (float*)(ws + OFF_EMA0);
  float*          ema1    = (float*)(ws + OFF_EMA1);
  float*          pnum    = (float*)(ws + OFF_PNUM);
  float*          pden    = (float*)(ws + OFF_PDEN);
  float*          pcnt    = (float*)(ws + OFF_PCNT);
  float*          term    = (float*)(ws + OFF_TERM);
  _Float16*       Xh      = (_Float16*)(ws + OFF_XH);
  float*          S0      = (float*)(ws + OFF_S0);
  float*          S1      = (float*)(ws + OFF_S1);
  unsigned char*  M0      = (unsigned char*)(ws + OFF_M0);
  unsigned char*  M1      = (unsigned char*)(ws + OFF_M1);

  dim3 gGemm(32, 128), bGemm(128);
  dim3 gM(128, 128), bM(16, 16);

  // ---- stage 1: last layer, full set -> subset selection ----
  const float* F7 = F + (size_t)(NLAYER - 1) * B_FULL * DFEAT;
  normalize_gather<<<256, 256, 0, stream>>>(F7, nullptr, nullptr, Xh);
  gemm_xxt<<<gGemm, bGemm, 0, stream>>>(Xh, S0);
  topk_rows<<<256, 256, 0, stream>>>(S0, labels, nullptr, topi, keep);
  build_M<<<gM, bM, 0, stream>>>(labels, nullptr, topi, keep, M0);
  row_any<<<256, 256, 0, stream>>>(M0, flags);
  compact<<<1, 1, 0, stream>>>(flags, labels, meta, idx_sub, lsub);

  // ---- stage 2: per-layer subset pipeline with sliding pair loss ----
  for (int l = 0; l < NLAYER; ++l) {
    float*         S = (l & 1) ? S1 : S0;
    unsigned char* M = (l & 1) ? M1 : M0;
    float*         E = (l & 1) ? ema1 : ema0;
    const float* Fl = F + (size_t)l * B_FULL * DFEAT;
    normalize_gather<<<256, 256, 0, stream>>>(Fl, idx_sub, meta, Xh);
    gemm_xxt<<<gGemm, bGemm, 0, stream>>>(Xh, S);
    topk_rows<<<256, 256, 0, stream>>>(S, lsub, meta, topi, keep);
    build_M<<<gM, bM, 0, stream>>>(lsub, meta, topi, keep, M);
    score_rows<<<256, 256, 0, stream>>>(S, lsub, meta, E);
    if (l > 0) {
      float*         Sp = (l & 1) ? S0 : S1;
      unsigned char* Mp = (l & 1) ? M0 : M1;
      float*         Ep = (l & 1) ? ema0 : ema1;
      pair_rows<<<B_FULL, 256, 0, stream>>>(Sp, S, Mp, M, Ep, meta, pnum, pden, pcnt);
      pair_reduce<<<1, 1, 0, stream>>>(pnum, pden, pcnt, term, l - 1);
    }
  }
  finalize<<<1, 1, 0, stream>>>(term, (float*)d_out);
}
